// Mamba2Block_12575664243475
// MI455X (gfx1250) — matrix operand order repaired and checked by reading, not compiled, not run on hardware
//
#include <hip/hip_runtime.h>
#include <hip/hip_bf16.h>
#include <cstdint>

// ---------------- Problem constants (match reference) ----------------
constexpr int BATCH   = 2;
constexpr int SEQ     = 2048;
constexpr int BL      = BATCH * SEQ;        // 4096 rows
constexpr int DMODEL  = 1024;
constexpr int DINNER  = 2048;
constexpr int NHEADS  = 32;
constexpr int HEADDIM = 64;
constexpr int DSTATE  = 128;
constexpr int CHUNK   = 256;
constexpr int NCHUNK  = SEQ / CHUNK;        // 8
constexpr int CONVD   = DINNER + 2 * DSTATE;      // 2304
constexpr int DINPROJ = 2 * DINNER + 2 * DSTATE + NHEADS; // 4384
constexpr float EPS   = 1e-5f;

typedef __bf16 bf16_t;
typedef __attribute__((ext_vector_type(16))) __bf16 v16bf;
typedef __attribute__((ext_vector_type(8)))  float  v8f;
typedef __attribute__((ext_vector_type(4)))  unsigned int v4u;
typedef __attribute__((ext_vector_type(8)))  int v8i;
typedef __attribute__((ext_vector_type(4)))  int v4i;

#define WMMA_BF16(a, b, c) \
  __builtin_amdgcn_wmma_f32_16x16x32_bf16(false, (a), false, (b), (short)0, (c), false, false)

// ---------------- TDM: 2D tensor tile -> LDS (CDNA5 Tensor Data Mover) ----------------
// D# layout per CDNA5 ISA.
// Loads a tile of (tile1 rows x tile0 elems) of 2-byte elements from a row-major
// tensor (row stride = stride0 elems) into LDS at byte offset lds_addr.
__device__ __forceinline__ void tdm_load_2d_bf16(unsigned lds_addr, const void* gaddr,
                                                 unsigned dim0, unsigned dim1,
                                                 unsigned tile0, unsigned tile1,
                                                 unsigned stride0) {
  unsigned long long ga = (unsigned long long)(uintptr_t)gaddr;
  v4u g0;
  g0[0] = 1u;                                         // count=1, user mode, no gather
  g0[1] = lds_addr;                                   // LDS byte address
  g0[2] = (unsigned)(ga & 0xffffffffu);               // global_addr[31:0]
  g0[3] = (unsigned)((ga >> 32) & 0x01ffffffu) | (2u << 30); // global_addr[56:32], type=2
  v8i g1;
  g1[0] = (int)(1u << 16);                            // workgroup_mask=0, data_size=1 (2B)
  g1[1] = (int)((dim0 & 0xffffu) << 16);              // tensor_dim0[15:0] in [31:16]
  g1[2] = (int)((dim0 >> 16) | ((dim1 & 0xffffu) << 16));   // dim0 hi | dim1 lo
  g1[3] = (int)((dim1 >> 16) | ((tile0 & 0xffffu) << 16));  // dim1 hi | tile_dim0
  g1[4] = (int)(tile1 & 0xffffu);                     // tile_dim1 ; tile_dim2=0
  g1[5] = (int)stride0;                               // tensor_dim0_stride[31:0]
  g1[6] = 0;                                          // stride0[47:32]=0 | stride1 lo=0
  g1[7] = 0;
  v4i gz = {0, 0, 0, 0};
#if defined(__clang_major__) && (__clang_major__ >= 23)
  v8i gz8 = {0, 0, 0, 0, 0, 0, 0, 0};
  __builtin_amdgcn_tensor_load_to_lds(g0, g1, gz, gz, gz8, 0);
#else
  __builtin_amdgcn_tensor_load_to_lds(g0, g1, gz, gz, 0);
#endif
}

// ---------------- WMMA fragment loaders (wave32, ISA 7.12.2) ----------------
// A 16x32 (MxK), source row-major with leading dim ld (in elements).
__device__ __forceinline__ v16bf load_a_rowmajor(const bf16_t* src, int ld) {
  int lane = threadIdx.x & 31;
  int r  = lane & 15;
  int kb = (lane >> 4) << 4;                // 0 or 16
  const bf16_t* p = src + (size_t)r * ld + kb;
  v16bf a;
#pragma unroll
  for (int i = 0; i < 8; ++i) a[i] = p[i];          // K = kb..kb+7
#pragma unroll
  for (int i = 0; i < 8; ++i) a[8 + i] = p[8 + i];  // K = kb+8..kb+15
  return a;
}

// A 16x32 where memory is K-major (stored [K][M], leading dim ld = M-stride).
__device__ __forceinline__ v16bf load_a_kmajor(const bf16_t* src, int ld) {
  int lane = threadIdx.x & 31;
  int r  = lane & 15;
  int kb = (lane >> 4) << 4;
  v16bf a;
#pragma unroll
  for (int i = 0; i < 8; ++i) a[i]     = src[(size_t)(kb + i) * ld + r];
#pragma unroll
  for (int i = 0; i < 8; ++i) a[8 + i] = src[(size_t)(kb + 8 + i) * ld + r];
  return a;
}

// B 32x16 (KxN) from an N-major source W[N][K] (row-major, ld = K-stride): B[k][n] = W[n][k].
__device__ __forceinline__ v16bf load_b_nmajor(const bf16_t* src, int ld) {
  int lane = threadIdx.x & 31;
  int n  = lane & 15;
  int kb = (lane >> 4) << 4;                // 0 or 16
  const bf16_t* p = src + (size_t)n * ld + kb;
  v16bf b;
#pragma unroll
  for (int i = 0; i < 16; ++i) b[i] = p[i];
  return b;
}

// B 32x16 (KxN) from a K-major source M[K][N] (row-major, ld = N-stride).
__device__ __forceinline__ v16bf load_b_kmajor(const bf16_t* src, int ld) {
  int lane = threadIdx.x & 31;
  int n  = lane & 15;
  int kb = (lane >> 4) << 4;
  const bf16_t* p = src + (size_t)kb * ld + n;
  v16bf b;
#pragma unroll
  for (int i = 0; i < 16; ++i) b[i] = p[(size_t)i * ld];
  return b;
}

// Store 16x16 f32 D tile to row-major dst with leading dim ld.
__device__ __forceinline__ void store_d(float* dst, int ld, const v8f& d) {
  int lane = threadIdx.x & 31;
  int n  = lane & 15;
  int mb = (lane >> 4) << 3;
#pragma unroll
  for (int v = 0; v < 8; ++v) dst[(size_t)(mb + v) * ld + n] = d[v];
}

// ---------------- Elementwise / prep kernels ----------------
__global__ void f32_to_bf16_kernel(const float* __restrict__ src, bf16_t* __restrict__ dst, size_t n) {
  size_t i = (size_t)blockIdx.x * 256 + threadIdx.x;
  if (i < n) dst[i] = (bf16_t)src[i];
}

// rmsnorm over DMODEL=1024, output bf16. 1 block per row, 256 threads * 4 elems.
__global__ void rmsnorm_in_kernel(const float* __restrict__ x, const float* __restrict__ w,
                                  bf16_t* __restrict__ out) {
  __shared__ float red[256];
  size_t row = blockIdx.x;
  const float* xr = x + row * DMODEL;
  float v[4]; float ss = 0.f;
#pragma unroll
  for (int i = 0; i < 4; ++i) { v[i] = xr[threadIdx.x + i * 256]; ss += v[i] * v[i]; }
  red[threadIdx.x] = ss; __syncthreads();
  for (int s = 128; s > 0; s >>= 1) { if ((int)threadIdx.x < s) red[threadIdx.x] += red[threadIdx.x + s]; __syncthreads(); }
  float scale = rsqrtf(red[0] * (1.f / DMODEL) + EPS);
  bf16_t* orow = out + row * DMODEL;
#pragma unroll
  for (int i = 0; i < 4; ++i) {
    int idx = threadIdx.x + i * 256;
    orow[idx] = (bf16_t)(v[i] * scale * w[idx]);
  }
}

// Depthwise causal conv (D_CONV=4) + bias + SiLU over xBC slice of zxbcdt.
__global__ void conv_silu_kernel(const float* __restrict__ zxbcdt, const float* __restrict__ cw,
                                 const float* __restrict__ cb, float* __restrict__ conv_out) {
  int ch = blockIdx.y * 256 + threadIdx.x;           // 0..2303
  long long bl = blockIdx.x;                          // 0..BL-1
  int l = (int)(bl & (SEQ - 1));
  float s = cb[ch];
#pragma unroll
  for (int k = 0; k < 4; ++k) {
    int li = l - 3 + k;
    if (li >= 0)
      s += cw[ch * 4 + k] * zxbcdt[(bl + k - 3) * (long long)DINPROJ + DINNER + ch];
  }
  s = s / (1.f + expf(-s));                           // SiLU
  conv_out[bl * (long long)CONVD + ch] = s;
}

// dt = softplus(dt + bias); a = dt * (-exp(A_log))
__global__ void dt_prep_kernel(const float* __restrict__ zxbcdt, const float* __restrict__ dt_bias,
                               const float* __restrict__ A_log,
                               float* __restrict__ dtproc, float* __restrict__ a_arr) {
  int i = blockIdx.x * 256 + threadIdx.x;             // over BL*NHEADS
  if (i >= BL * NHEADS) return;
  int h = i & (NHEADS - 1);
  int bl = i >> 5;
  float v = zxbcdt[(size_t)bl * DINPROJ + (2 * DINNER + 2 * DSTATE) + h] + dt_bias[h];
  float sp = (v > 20.f) ? v : log1pf(expf(v));
  dtproc[i] = sp;
  a_arr[i] = -expf(A_log[h]) * sp;
}

// Per (b,c): 32 threads (one per head) cumsum a over the chunk.
__global__ void cumsum_kernel(const float* __restrict__ a_arr,
                              float* __restrict__ acs, float* __restrict__ csum) {
  int bc = blockIdx.x; int b = bc >> 3; int c = bc & (NCHUNK - 1);
  int h = threadIdx.x;
  float run = 0.f;
  size_t base  = ((size_t)b * SEQ + (size_t)c * CHUNK);
  size_t obase = (((size_t)b * NHEADS + h) * NCHUNK + c) * CHUNK;
  for (int i = 0; i < CHUNK; ++i) {
    run += a_arr[(base + i) * NHEADS + h];
    acs[obase + i] = run;
  }
  csum[((size_t)b * NHEADS + h) * NCHUNK + c] = run;
}

// xdt = bf16(xs * dt) ; B,C chunks -> bf16
__global__ void scale_convert_kernel(const float* __restrict__ conv_out, const float* __restrict__ dtproc,
                                     bf16_t* __restrict__ xdt, bf16_t* __restrict__ Bb, bf16_t* __restrict__ Cb) {
  size_t bl = blockIdx.x;
  const float* r = conv_out + bl * CONVD;
#pragma unroll
  for (int i = 0; i < 8; ++i) {
    int idx = threadIdx.x + i * 256;                  // 0..2047
    int h = idx >> 6;
    xdt[bl * DINNER + idx] = (bf16_t)(r[idx] * dtproc[bl * NHEADS + h]);
  }
  int n = threadIdx.x & 127;
  if (threadIdx.x < 128) Bb[bl * DSTATE + n] = (bf16_t)r[DINNER + n];
  else                   Cb[bl * DSTATE + n] = (bf16_t)r[DINNER + DSTATE + n];
}

// ---------------- WMMA GEMM: out[M,N] = A[M,K](bf16) * W[N,K](bf16)^T (+res) ----------------
// W tiles staged via the Tensor Data Mover (double buffered), synced with TENSORcnt.
template <bool ADD_RES>
__global__ void wmma_gemm_kernel(const bf16_t* __restrict__ A, const bf16_t* __restrict__ W,
                                 const float* __restrict__ res, float* __restrict__ out,
                                 int M, int N, int K) {
  __shared__ bf16_t Bs[2][32 * 32];                   // ping-pong W tile [n][k]
  const int nblk = blockIdx.x * 32;
  const int mblk = blockIdx.y * 128;
  const int wave = threadIdx.x >> 5;
  const int m0 = mblk + wave * 16;
  v8f acc0 = {}, acc1 = {};

  const bf16_t* Wtile = W + (size_t)nblk * K;
  if (wave == 0) {
    tdm_load_2d_bf16((unsigned)(uintptr_t)&Bs[0][0], Wtile, (unsigned)K, (unsigned)N,
                     32u, 32u, (unsigned)K);
    __builtin_amdgcn_s_wait_tensorcnt(0);
  }
  __syncthreads();

  int buf = 0;
  for (int k0 = 0; k0 < K; k0 += 32) {
    if (wave == 0 && (k0 + 32) < K) {                 // async prefetch of next W tile
      tdm_load_2d_bf16((unsigned)(uintptr_t)&Bs[buf ^ 1][0], Wtile + k0 + 32,
                       (unsigned)K, (unsigned)N, 32u, 32u, (unsigned)K);
    }
    // prefetch next A tile into caches
    if ((k0 + 32) < K)
      __builtin_prefetch(A + (size_t)m0 * K + k0 + 32, 0, 0);

    v16bf a  = load_a_rowmajor(A + (size_t)m0 * K + k0, K);
    v16bf b0 = load_b_nmajor(&Bs[buf][0], 32);
    v16bf b1 = load_b_nmajor(&Bs[buf][16 * 32], 32);
    acc0 = WMMA_BF16(a, b0, acc0);
    acc1 = WMMA_BF16(a, b1, acc1);

    if (wave == 0) __builtin_amdgcn_s_wait_tensorcnt(0);
    __syncthreads();
    buf ^= 1;
  }

  int lane = threadIdx.x & 31;
  int n = lane & 15, mb = (lane >> 4) << 3;
  size_t ob = (size_t)(m0 + mb) * N + nblk;
#pragma unroll
  for (int v = 0; v < 8; ++v) {
    float r0 = acc0[v], r1 = acc1[v];
    if (ADD_RES) {
      r0 += res[ob + (size_t)v * N + n];
      r1 += res[ob + (size_t)v * N + 16 + n];
    }
    out[ob + (size_t)v * N + n]      = r0;
    out[ob + (size_t)v * N + 16 + n] = r1;
  }
}

// ---------------- SSD: diagonal block + chunk states (per b,chunk,head) ----------------
constexpr int LDG = CHUNK + 8;                        // padded bf16 LDS stride

__global__ void ssd_diag_kernel(const bf16_t* __restrict__ Cb, const bf16_t* __restrict__ Bb,
                                const bf16_t* __restrict__ xdt, const float* __restrict__ acs,
                                float* __restrict__ ydiag, float* __restrict__ states) {
  extern __shared__ char smem[];
  bf16_t* Gs    = (bf16_t*)smem;                      // [CHUNK][LDG] masked G in bf16
  float*  acs_s = (float*)(smem + (size_t)CHUNK * LDG * sizeof(bf16_t));
  float*  dec_s = acs_s + CHUNK;

  const int h = blockIdx.x, c = blockIdx.y, b = blockIdx.z;
  const int tid = threadIdx.x, wave = tid >> 5;
  const size_t lbase = (size_t)b * SEQ + (size_t)c * CHUNK;

  float av = acs[(((size_t)b * NHEADS + h) * NCHUNK + c) * CHUNK + tid];
  acs_s[tid] = av;
  __syncthreads();
  dec_s[tid] = expf(acs_s[CHUNK - 1] - av);
  __syncthreads();

  // Phase 1: G = C * B^T, masked by Lm = tril(exp(acs[l]-acs[s])), stored bf16 in LDS.
  for (int t = wave; t < 256; t += 8) {               // 16x16 grid of 16x16 tiles
    int ti = t >> 4, tj = t & 15;
    v8f acc = {};
    const bf16_t* Ap = Cb + (lbase + (size_t)ti * 16) * DSTATE;
    const bf16_t* Bp = Bb + (lbase + (size_t)tj * 16) * DSTATE;
#pragma unroll
    for (int k0 = 0; k0 < DSTATE; k0 += 32) {
      v16bf a  = load_a_rowmajor(Ap + k0, DSTATE);
      v16bf bm = load_b_nmajor(Bp + k0, DSTATE);
      acc = WMMA_BF16(a, bm, acc);
    }
    int lane = tid & 31, n = lane & 15, mb = (lane >> 4) << 3;
    int s = tj * 16 + n;
#pragma unroll
    for (int v = 0; v < 8; ++v) {
      int l = ti * 16 + mb + v;
      float g = (l >= s) ? acc[v] * expf(acs_s[l] - acs_s[s]) : 0.f;
      Gs[(size_t)l * LDG + s] = (bf16_t)g;
    }
  }
  __syncthreads();

  // Phase 2: Y_diag = (G∘L) @ xd   (256x64, K=256)
  for (int t = wave; t < 64; t += 8) {
    int ti = t >> 2, tp = t & 3;
    v8f acc = {};
#pragma unroll
    for (int s0 = 0; s0 < CHUNK; s0 += 32) {
      v16bf a  = load_a_rowmajor(Gs + (size_t)ti * 16 * LDG + s0, LDG);
      v16bf bm = load_b_kmajor(xdt + (lbase + s0) * DINNER + h * HEADDIM + tp * 16, DINNER);
      acc = WMMA_BF16(a, bm, acc);
    }
    store_d(ydiag + (lbase + (size_t)ti * 16) * DINNER + h * HEADDIM + tp * 16, DINNER, acc);
  }

  // Phase 3: chunk state S[n][p] = sum_l B[l][n] * decay[l] * xd[l][p]   (128x64, K=256)
  for (int t = wave; t < 32; t += 8) {
    int tn = t >> 2, tp = t & 3;
    v8f acc = {};
#pragma unroll
    for (int k0 = 0; k0 < CHUNK; k0 += 32) {
      v16bf a = load_a_kmajor(Bb + (lbase + k0) * DSTATE + tn * 16, DSTATE);
      int lane = tid & 31, n = lane & 15, kb = (lane >> 4) << 4;
      v16bf bm;
#pragma unroll
      for (int i = 0; i < 16; ++i) {
        int l = k0 + kb + i;
        float xv = (float)xdt[(lbase + l) * DINNER + h * HEADDIM + tp * 16 + n];
        bm[i] = (bf16_t)(xv * dec_s[l]);
      }
      acc = WMMA_BF16(a, bm, acc);
    }
    store_d(states + ((((size_t)b * NCHUNK + c) * NHEADS + h) * DSTATE + tn * 16) * HEADDIM + tp * 16,
            HEADDIM, acc);
  }
}

// ---------------- Inter-chunk recurrence (sequential over NCHUNK=8) ----------------
__global__ void chunk_scan_kernel(const float* __restrict__ states, const float* __restrict__ csum,
                                  bf16_t* __restrict__ init_bf) {
  int bh = blockIdx.x;                                // b*NHEADS + h
  float r[32];
#pragma unroll
  for (int i = 0; i < 32; ++i) r[i] = 0.f;
  for (int c = 0; c < NCHUNK; ++c) {
    size_t base = (((size_t)(bh >> 5) * NCHUNK + c) * NHEADS + (bh & 31)) * (DSTATE * HEADDIM);
#pragma unroll
    for (int i = 0; i < 32; ++i) init_bf[base + threadIdx.x + i * 256] = (bf16_t)r[i];
    float g = expf(csum[(size_t)bh * NCHUNK + c]);
#pragma unroll
    for (int i = 0; i < 32; ++i) r[i] = r[i] * g + states[base + threadIdx.x + i * 256];
  }
}

// ---------------- SSD: off-diagonal Y_off + combine + D skip ----------------
__global__ void ssd_yoff_kernel(const bf16_t* __restrict__ Cb, const bf16_t* __restrict__ init_bf,
                                const float* __restrict__ acs, const float* __restrict__ ydiag,
                                const float* __restrict__ conv_out, const float* __restrict__ Dp,
                                float* __restrict__ y) {
  __shared__ float acs_s[CHUNK];
  const int h = blockIdx.x, c = blockIdx.y, b = blockIdx.z;
  const int tid = threadIdx.x, wave = tid >> 5;
  const size_t lbase = (size_t)b * SEQ + (size_t)c * CHUNK;
  acs_s[tid] = acs[(((size_t)b * NHEADS + h) * NCHUNK + c) * CHUNK + tid];
  __syncthreads();
  const float Dh = Dp[h];
  const bf16_t* Sp = init_bf + (((size_t)b * NCHUNK + c) * NHEADS + h) * (DSTATE * HEADDIM); // [n][p]

  for (int t = wave; t < 64; t += 8) {
    int ti = t >> 2, tp = t & 3;
    v8f acc = {};
#pragma unroll
    for (int k0 = 0; k0 < DSTATE; k0 += 32) {
      v16bf a  = load_a_rowmajor(Cb + (lbase + (size_t)ti * 16) * DSTATE + k0, DSTATE);
      v16bf bm = load_b_kmajor(Sp + (size_t)k0 * HEADDIM + tp * 16, HEADDIM);
      acc = WMMA_BF16(a, bm, acc);
    }
    int lane = tid & 31, n = lane & 15, mb = (lane >> 4) << 3;
#pragma unroll
    for (int v = 0; v < 8; ++v) {
      int l = ti * 16 + mb + v;
      size_t row = lbase + l;
      int col = h * HEADDIM + tp * 16 + n;
      float xs = conv_out[row * CONVD + col];
      y[row * DINNER + col] = acc[v] * expf(acs_s[l]) + ydiag[row * DINNER + col] + Dh * xs;
    }
  }
}

// ---------------- Gated rmsnorm over DINNER=2048 -> bf16 ----------------
__global__ void gate_norm_kernel(const float* __restrict__ y, const float* __restrict__ zxbcdt,
                                 const float* __restrict__ gw, bf16_t* __restrict__ out) {
  __shared__ float red[256];
  size_t row = blockIdx.x;
  const float* yr = y + row * DINNER;
  const float* zr = zxbcdt + row * DINPROJ;           // z = first DINNER columns
  float t[8]; float ss = 0.f;
#pragma unroll
  for (int i = 0; i < 8; ++i) {
    int idx = threadIdx.x + i * 256;
    float zv = zr[idx];
    float val = yr[idx] * (zv / (1.f + expf(-zv)));
    t[i] = val; ss += val * val;
  }
  red[threadIdx.x] = ss; __syncthreads();
  for (int s = 128; s > 0; s >>= 1) { if ((int)threadIdx.x < s) red[threadIdx.x] += red[threadIdx.x + s]; __syncthreads(); }
  float scale = rsqrtf(red[0] * (1.f / DINNER) + EPS);
#pragma unroll
  for (int i = 0; i < 8; ++i) {
    int idx = threadIdx.x + i * 256;
    out[row * DINNER + idx] = (bf16_t)(t[i] * scale * gw[idx]);
  }
}

// ---------------- Host launcher ----------------
extern "C" void kernel_launch(void* const* d_in, const int* in_sizes, int n_in,
                              void* d_out, int out_size, void* d_ws, size_t ws_size,
                              hipStream_t stream) {
  const float* x        = (const float*)d_in[0];
  const float* norm_w   = (const float*)d_in[1];
  const float* in_w     = (const float*)d_in[2];
  const float* conv_w   = (const float*)d_in[3];
  const float* conv_b   = (const float*)d_in[4];
  const float* dt_bias  = (const float*)d_in[5];
  const float* A_log    = (const float*)d_in[6];
  const float* Dp       = (const float*)d_in[7];
  const float* gnorm_w  = (const float*)d_in[8];
  const float* out_w    = (const float*)d_in[9];
  float* out = (float*)d_out;

  char* ws = (char*)d_ws;
  size_t off = 0;
  auto alloc = [&](size_t bytes) -> char* {
    off = (off + 255) & ~(size_t)255;
    char* p = ws + off; off += bytes; return p;
  };

  bf16_t* in_w_bf  = (bf16_t*)alloc((size_t)DINPROJ * DMODEL * 2);
  bf16_t* out_w_bf = (bf16_t*)alloc((size_t)DMODEL * DINNER * 2);
  bf16_t* u_bf     = (bf16_t*)alloc((size_t)BL * DMODEL * 2);
  float*  zxbcdt   = (float*)alloc((size_t)BL * DINPROJ * 4);
  float*  conv_out = (float*)alloc((size_t)BL * CONVD * 4);
  float*  dtproc   = (float*)alloc((size_t)BL * NHEADS * 4);
  float*  a_arr    = (float*)alloc((size_t)BL * NHEADS * 4);
  float*  acs      = (float*)alloc((size_t)BATCH * NHEADS * NCHUNK * CHUNK * 4);
  float*  csum     = (float*)alloc((size_t)BATCH * NHEADS * NCHUNK * 4);
  bf16_t* xdt      = (bf16_t*)alloc((size_t)BL * DINNER * 2);
  bf16_t* Bb       = (bf16_t*)alloc((size_t)BL * DSTATE * 2);
  bf16_t* Cb       = (bf16_t*)alloc((size_t)BL * DSTATE * 2);
  float*  ydiag    = (float*)alloc((size_t)BL * DINNER * 4);
  float*  states   = (float*)alloc((size_t)BATCH * NCHUNK * NHEADS * DSTATE * HEADDIM * 4);
  bf16_t* init_bf  = (bf16_t*)alloc((size_t)BATCH * NCHUNK * NHEADS * DSTATE * HEADDIM * 2);
  float*  yfull    = (float*)alloc((size_t)BL * DINNER * 4);
  bf16_t* yn_bf    = (bf16_t*)alloc((size_t)BL * DINNER * 2);

  // 1) weight conversions
  {
    size_t n1 = (size_t)DINPROJ * DMODEL;
    f32_to_bf16_kernel<<<(unsigned)((n1 + 255) / 256), 256, 0, stream>>>(in_w, in_w_bf, n1);
    size_t n2 = (size_t)DMODEL * DINNER;
    f32_to_bf16_kernel<<<(unsigned)((n2 + 255) / 256), 256, 0, stream>>>(out_w, out_w_bf, n2);
  }
  // 2) input rmsnorm -> bf16
  rmsnorm_in_kernel<<<BL, 256, 0, stream>>>(x, norm_w, u_bf);
  // 3) in_proj GEMM: [BL,DINPROJ] = u_bf[BL,DMODEL] * in_w[DINPROJ,DMODEL]^T
  wmma_gemm_kernel<false><<<dim3(DINPROJ / 32, BL / 128), 256, 0, stream>>>(
      u_bf, in_w_bf, nullptr, zxbcdt, BL, DINPROJ, DMODEL);
  // 4) depthwise conv + SiLU
  conv_silu_kernel<<<dim3(BL, CONVD / 256), 256, 0, stream>>>(zxbcdt, conv_w, conv_b, conv_out);
  // 5) dt softplus + a
  dt_prep_kernel<<<(BL * NHEADS + 255) / 256, 256, 0, stream>>>(zxbcdt, dt_bias, A_log, dtproc, a_arr);
  // 6) per-chunk cumulative sums
  cumsum_kernel<<<BATCH * NCHUNK, 32, 0, stream>>>(a_arr, acs, csum);
  // 7) xdt / B / C -> bf16
  scale_convert_kernel<<<BL, 256, 0, stream>>>(conv_out, dtproc, xdt, Bb, Cb);
  // 8) SSD diagonal + chunk states (WMMA)
  {
    size_t shmem = (size_t)CHUNK * LDG * sizeof(bf16_t) + 2 * CHUNK * sizeof(float);
    ssd_diag_kernel<<<dim3(NHEADS, NCHUNK, BATCH), 256, shmem, stream>>>(
        Cb, Bb, xdt, acs, ydiag, states);
  }
  // 9) inter-chunk recurrence -> init states (bf16)
  chunk_scan_kernel<<<BATCH * NHEADS, 256, 0, stream>>>(states, csum, init_bf);
  // 10) off-diagonal contribution + combine + D skip (WMMA)
  ssd_yoff_kernel<<<dim3(NHEADS, NCHUNK, BATCH), 256, 0, stream>>>(
      Cb, init_bf, acs, ydiag, conv_out, Dp, yfull);
  // 11) gated rmsnorm -> bf16
  gate_norm_kernel<<<BL, 256, 0, stream>>>(yfull, zxbcdt, gnorm_w, yn_bf);
  // 12) out_proj GEMM + residual
  wmma_gemm_kernel<true><<<dim3(DMODEL / 32, BL / 128), 256, 0, stream>>>(
      yn_bf, out_w_bf, x, out, BL, DMODEL, DINNER);
  (void)in_sizes; (void)n_in; (void)out_size; (void)ws_size;
}
